// SelfAttentionWith1DRelativePos_23811298689413
// MI455X (gfx1250) — matrix operand order repaired and checked by reading, not compiled, not run on hardware
//
#include <hip/hip_runtime.h>

// ---------------------------------------------------------------------------
// SelfAttention with 1D relative position — MI455X (gfx1250) WMMA pipeline
//   B=4, L=1024, C=512, H=8, hf=64 (q/k 32ch, v 64ch), groups=32, eps=1e-4
// Round 2: fragment-major packed operands (b128 loads), TJ=32 attention step
// with full-K P@V WMMAs and A-fragment-ordered P staging in LDS.
// ---------------------------------------------------------------------------

typedef __attribute__((ext_vector_type(16))) __bf16        v16bf;
typedef __attribute__((ext_vector_type(8)))  float         v8f;
typedef __attribute__((ext_vector_type(4)))  unsigned int  u32x4;

#define B_   4
#define L_   1024
#define C_   512
#define H_   8
#define C2_  1024
#define DQ   32
#define DV   64
#define TL_  2047      // rel_emb second dim (2*L - 1)
#define GEPS 1.0e-4f
#define EBS  49        // Eband row stride (47-wide band, padded)

static __device__ __forceinline__ unsigned short f2bf(float f) {
  unsigned int x = __float_as_uint(f);
  unsigned int r = (x + 0x7FFFu + ((x >> 16) & 1u)) >> 16;  // RNE
  return (unsigned short)r;
}
static __device__ __forceinline__ float bf2f(unsigned short u) {
  return __uint_as_float(((unsigned int)u) << 16);
}

union Frag16 {
  v16bf v;
  unsigned short h[16];
  u32x4 q4[2];
};

// A-fragment inverse mapping (16x32 bf16, wave32):
//   element (row m, k) lives in lane  m + 16*((k>>4)&1), slot (k&15)
static __device__ __forceinline__ int aLane(int m, int k) { return m + 16 * ((k >> 4) & 1); }
static __device__ __forceinline__ int aSlot(int k)        { return (k & 15); }

// ---------------------------------------------------------------------------
// Pack x -> A-fragment-major bf16:  pA[((tm*16 + kt)*32 + lane)*16 + i]
//   tm: 16-row tile (B*L/16), kt: 32-wide K tile (C/32)
// ---------------------------------------------------------------------------
__global__ void k_pack_a(const float* __restrict__ x, unsigned short* __restrict__ pA) {
  const int idx = blockIdx.x * blockDim.x + threadIdx.x;
  if (idx >= B_ * L_ * C_) return;
  const int i  = idx & 15;
  const int l  = (idx >> 4) & 31;
  const int kt = (idx >> 9) & 15;           // C/32 = 16
  const int tm = idx >> 13;
  const int row = tm * 16 + (l & 15);
  const int K = kt * 32 + (l >> 4) * 16 + i;
  pA[idx] = f2bf(x[row * C_ + K]);
}

// ---------------------------------------------------------------------------
// Pack W_qkv -> B-fragment-major bf16: pB[((kt*64 + tn)*32 + lane)*16 + i]
//   B-frag: col = lane&15, K = (lane>>4)*16 + i
// ---------------------------------------------------------------------------
__global__ void k_pack_b(const float* __restrict__ W, unsigned short* __restrict__ pB) {
  const int idx = blockIdx.x * blockDim.x + threadIdx.x;
  if (idx >= C_ * C2_) return;
  const int i  = idx & 15;
  const int l  = (idx >> 4) & 31;
  const int tn = (idx >> 9) & 63;           // C2/16 = 64
  const int kt = idx >> 15;                 // C/32 = 16
  const int kk = kt * 32 + (l >> 4) * 16 + i;
  const int col = tn * 16 + (l & 15);
  pB[idx] = f2bf(W[kk * C2_ + col]);
}

// ---------------------------------------------------------------------------
// QKV projection GEMM: [B*L, C] x [C, C2] -> [B*L, C2] fp32.
// 8 waves/block; each wave computes a 16x64 strip: A-frag reused over 4 B-frags,
// 4 WMMAs per K-step, all operand fetches are global_load_b128 pairs.
// ---------------------------------------------------------------------------
__global__ void __launch_bounds__(256)
k_qkv_gemm(const unsigned short* __restrict__ pA,
           const unsigned short* __restrict__ pB,
           float* __restrict__ qkv) {
  const int lane = threadIdx.x & 31;
  const int wave = threadIdx.x >> 5;
  const int strip = blockIdx.x * 8 + wave;  // 4096 strips (256 row x 16 col)
  const int ts = strip & 15;
  const int tm = strip >> 4;
  const int m0 = tm * 16, n0 = ts * 64;

  const u32x4* pa4 = (const u32x4*)pA;
  const u32x4* pb4 = (const u32x4*)pB;

  v8f acc[4] = {};
  for (int kt = 0; kt < C_ / 32; ++kt) {
    Frag16 a;
    const int fa = (tm * 16 + kt) * 32 + lane;
    a.q4[0] = pa4[fa * 2 + 0];
    a.q4[1] = pa4[fa * 2 + 1];
    __builtin_prefetch(&pa4[(fa + 32) * 2], 0, 1);   // next K-step A tile
#pragma unroll
    for (int f = 0; f < 4; ++f) {
      Frag16 b;
      const int fb = (kt * 64 + (ts * 4 + f)) * 32 + lane;
      b.q4[0] = pb4[fb * 2 + 0];
      b.q4[1] = pb4[fb * 2 + 1];
      acc[f] = __builtin_amdgcn_wmma_f32_16x16x32_bf16(false, a.v, false, b.v,
                                                       (short)0, acc[f], false, false);
    }
  }
  const int rbase = (lane >> 4) * 8;        // C-frag: M = r + 8*half
  const int colw = lane & 15;
#pragma unroll
  for (int f = 0; f < 4; ++f)
#pragma unroll
    for (int r = 0; r < 8; ++r)
      qkv[(m0 + rbase + r) * C2_ + n0 + f * 16 + colw] = acc[f][r];
}

// ---------------------------------------------------------------------------
// GroupNorm statistics: one block per (b, group); 32 groups, N = L * gs.
// ---------------------------------------------------------------------------
__global__ void __launch_bounds__(256)
k_gn_stats(const float* __restrict__ x, float* __restrict__ stats,
           int Ctot, int gs) {
  const int b = blockIdx.x >> 5;
  const int g = blockIdx.x & 31;
  const int N = L_ * gs;
  float s = 0.f, ss = 0.f;
  for (int e = threadIdx.x; e < N; e += 256) {
    const int l = e / gs, cc = e % gs;
    const float v = x[(b * L_ + l) * Ctot + g * gs + cc];
    s += v; ss += v * v;
  }
  __shared__ float sh[256], sh2[256];
  sh[threadIdx.x] = s; sh2[threadIdx.x] = ss;
  __syncthreads();
  for (int o = 128; o > 0; o >>= 1) {
    if (threadIdx.x < o) {
      sh[threadIdx.x]  += sh[threadIdx.x + o];
      sh2[threadIdx.x] += sh2[threadIdx.x + o];
    }
    __syncthreads();
  }
  if (threadIdx.x == 0) {
    const float mean = sh[0] / (float)N;
    const float var  = sh2[0] / (float)N - mean * mean;
    stats[blockIdx.x * 2 + 0] = mean;
    stats[blockIdx.x * 2 + 1] = rsqrtf(var + GEPS);
  }
}

// ---------------------------------------------------------------------------
// Normalize QKV, split into:
//   qT,kT: [bh][L][32] bf16 (K B-frag loads are 16-contiguous in d)
//   vP:    V packed in B-fragment order per (bh, 32-j tile, 16-d chunk):
//          vP[(((bh*32 + jt)*4 + f)*32 + lane)*16 + i]
//          holds V at j = jt*32 + (lane>>4)*16 + i, d = f*16 + (lane&15)
// ---------------------------------------------------------------------------
__global__ void k_qkv_norm_split(const float* __restrict__ qkv,
                                 const float* __restrict__ stats,
                                 const float* __restrict__ scale,
                                 const float* __restrict__ bias,
                                 unsigned short* __restrict__ qT,
                                 unsigned short* __restrict__ kT,
                                 unsigned short* __restrict__ vP) {
  const int idx = blockIdx.x * blockDim.x + threadIdx.x;
  if (idx >= B_ * L_ * C2_) return;
  const int c2 = idx & (C2_ - 1);
  const int l  = (idx >> 10) & (L_ - 1);
  const int b  = idx >> 20;
  const int g  = c2 >> 5;
  const float mean = stats[(b * 32 + g) * 2 + 0];
  const float rstd = stats[(b * 32 + g) * 2 + 1];
  const float v = (qkv[idx] - mean) * rstd * scale[c2] + bias[c2];
  const int h = c2 >> 7, cc = c2 & 127;
  const int bh = b * H_ + h;
  const unsigned short bv = f2bf(v);
  if (cc < 32) {
    qT[(bh * L_ + l) * DQ + cc] = bv;
  } else if (cc < 64) {
    kT[(bh * L_ + l) * DQ + (cc - 32)] = bv;
  } else {
    const int dd = cc - 64;
    const int jt = l >> 5, kk = l & 31;
    const int lane = (kk >> 4) * 16 + (dd & 15);
    const int f = dd >> 4, i = kk & 15;
    vP[((((bh * 32 + jt) * 4 + f) * 32 + lane) << 4) + i] = bv;
  }
}

// ---------------------------------------------------------------------------
// Flash attention with 1D relative position.  One wave per (b,h,16-row i-tile).
// Per 32-col j-step:
//   2 WMMA (Q^T K, two 16-col subtiles) + VALU rel-logit band (tl = ii-jj+31)
//   online softmax -> P staged directly in A-fragment order in LDS
//   4 WMMA P@V (full K=32, packed b128 V fragments) + VALU P@rpe_v band
// ---------------------------------------------------------------------------
__global__ void __launch_bounds__(32)
k_attention(const unsigned short* __restrict__ qT,
            const unsigned short* __restrict__ kT,
            const unsigned short* __restrict__ vP,
            const float* __restrict__ E,              // [128][2047]
            float* __restrict__ opre) {               // [B][L][C]
  const int lane = threadIdx.x;
  const int i0 = blockIdx.x * 16;
  const int h  = blockIdx.y;
  const int b  = blockIdx.z;
  const int bh = b * H_ + h;
  const unsigned short* q = qT + (size_t)bh * L_ * DQ;
  const unsigned short* k = kT + (size_t)bh * L_ * DQ;

  __shared__ float Eband[128 * EBS];             // 25 KB band, stride 49
  __shared__ float qs[16 * 32];
  __shared__ float ks[16 * 32];
  __shared__ __align__(16) unsigned short PstA[32 * 16];  // P in A-frag order

  for (int e = lane; e < 16 * 32; e += 32) {
    qs[e] = bf2f(q[(i0 + (e >> 5)) * DQ + (e & 31)]);
    ks[e] = bf2f(k[(i0 + (e >> 5)) * DQ + (e & 31)]);
  }

  // A-fragment of Q^T tile (16 x K=32), loaded once (scalar gather, cold).
  Frag16 aq;
  {
    const int row = i0 + (lane & 15);
    const int kh = (lane >> 4) * 16;
#pragma unroll
    for (int i = 0; i < 16; ++i)
      aq.h[i] = q[row * DQ + kh + i];
  }
  __syncthreads();

  const int half = lane >> 4;
  const int colw = lane & 15;
  float m[8], lsum[8];
  v8f O[4] = {};
#pragma unroll
  for (int r = 0; r < 8; ++r) { m[r] = -3.0e38f; lsum[r] = 0.f; }

  for (int j0 = 0; j0 < L_; j0 += 32) {
    const int jt = j0 >> 5;

    // ---- Q^T K for two 16-col subtiles (b128 fragment loads from kT) ----
    Frag16 bk0, bk1;
    {
      const int jc0 = j0 + colw;
      const int jc1 = j0 + 16 + colw;
      const u32x4* kp0 = (const u32x4*)(k + jc0 * DQ + half * 16);
      const u32x4* kp1 = (const u32x4*)(k + jc1 * DQ + half * 16);
      bk0.q4[0] = kp0[0]; bk0.q4[1] = kp0[1];
      bk1.q4[0] = kp1[0]; bk1.q4[1] = kp1[1];
    }
    v8f z = {};
    v8f S0 = __builtin_amdgcn_wmma_f32_16x16x32_bf16(false, aq.v, false, bk0.v,
                                                     (short)0, z, false, false);
    v8f S1 = __builtin_amdgcn_wmma_f32_16x16x32_bf16(false, aq.v, false, bk1.v,
                                                     (short)0, z, false, false);

    // ---- stage rel_emb band: rows 0..127, t in [tbase, tbase+47] ----
    const int tbase = i0 - j0 + 992;   // (i0 - (j0+31)) + (L-1); >= 0 always
    __syncthreads();                   // previous iter done with Eband/PstA
    for (int e = lane; e < 128 * 48; e += 32) {
      const int er = e / 48, ec = e % 48;
      const int t = tbase + ec;
      Eband[er * EBS + ec] = (t >= 0 && t < TL_) ? E[er * TL_ + t] : 0.f;
    }
    __syncthreads();

    // ---- relative-position logits (VALU): tl = ii - jj + 31 ----
    float Sv0[8], Sv1[8];
#pragma unroll
    for (int r = 0; r < 8; ++r) {
      const int ii = r + half * 8;
      const int tl0 = ii - colw + 31;          // subtile 0: jj = colw
      const int tl1 = ii - colw + 15;          // subtile 1: jj = colw + 16
      float a0 = 0.f, a1 = 0.f;
#pragma unroll
      for (int d = 0; d < 32; ++d) {
        const float qv = qs[ii * 32 + d], kv = ks[ii * 32 + d];
        a0 += qv * Eband[(32 + d) * EBS + tl0] + kv * Eband[d * EBS + tl0];
        a1 += qv * Eband[(32 + d) * EBS + tl1] + kv * Eband[d * EBS + tl1];
      }
      Sv0[r] = S0[r] + a0;
      Sv1[r] = S1[r] + a1;
    }

    // ---- online softmax over both subtiles ----
    float sc[8], P0[8], P1[8];
#pragma unroll
    for (int r = 0; r < 8; ++r) {
      float rmax = fmaxf(Sv0[r], Sv1[r]);
      rmax = fmaxf(rmax, __shfl_xor(rmax, 1, 32));
      rmax = fmaxf(rmax, __shfl_xor(rmax, 2, 32));
      rmax = fmaxf(rmax, __shfl_xor(rmax, 4, 32));
      rmax = fmaxf(rmax, __shfl_xor(rmax, 8, 32));
      const float mn = fmaxf(m[r], rmax);
      sc[r] = __expf(m[r] - mn);
      P0[r] = __expf(Sv0[r] - mn);
      P1[r] = __expf(Sv1[r] - mn);
      float rs = P0[r] + P1[r];
      rs += __shfl_xor(rs, 1, 32);
      rs += __shfl_xor(rs, 2, 32);
      rs += __shfl_xor(rs, 4, 32);
      rs += __shfl_xor(rs, 8, 32);
      lsum[r] = lsum[r] * sc[r] + rs;
      m[r] = mn;
    }
#pragma unroll
    for (int r = 0; r < 8; ++r) {
      O[0][r] *= sc[r]; O[1][r] *= sc[r]; O[2][r] *= sc[r]; O[3][r] *= sc[r];
    }

    // ---- stage P directly in A-fragment order ----
#pragma unroll
    for (int r = 0; r < 8; ++r) {
      const int ii = r + half * 8;
      PstA[aLane(ii, colw)      * 16 + aSlot(colw)]      = f2bf(P0[r]);
      PstA[aLane(ii, colw + 16) * 16 + aSlot(colw + 16)] = f2bf(P1[r]);
    }
    __syncthreads();

    Frag16 pa;                                   // 2x ds_load_b128
    {
      const u32x4* pp = (const u32x4*)&PstA[lane * 16];
      pa.q4[0] = pp[0];
      pa.q4[1] = pp[1];
    }

    // ---- O += P @ V (full K=32, packed b128 V fragments) ----
#pragma unroll
    for (int f = 0; f < 4; ++f) {
      Frag16 bv;
      const u32x4* vp = (const u32x4*)&vP[((((bh * 32 + jt) * 4 + f) * 32 + lane) << 4)];
      bv.q4[0] = vp[0];
      bv.q4[1] = vp[1];
      O[f] = __builtin_amdgcn_wmma_f32_16x16x32_bf16(false, pa.v, false, bv.v,
                                                     (short)0, O[f], false, false);
    }

    // ---- O += P @ rpe_v (VALU band) ----
#pragma unroll
    for (int f = 0; f < 4; ++f) {
      const int dcol = f * 16 + colw;
#pragma unroll
      for (int r = 0; r < 8; ++r) {
        const int ii = r + half * 8;
        float acc = 0.f;
#pragma unroll
        for (int jj = 0; jj < 32; ++jj)
          acc += bf2f(PstA[aLane(ii, jj) * 16 + aSlot(jj)]) *
                 Eband[(64 + dcol) * EBS + (ii - jj + 31)];
        O[f][r] += acc;
      }
    }
  }

  // ---- epilogue: normalize by row sums, write (b, i, h*64 + d) fp32 ----
#pragma unroll
  for (int r = 0; r < 8; ++r) {
    const int ii = r + half * 8;
    const float inv = 1.0f / lsum[r];
    const int gi = i0 + ii;
#pragma unroll
    for (int f = 0; f < 4; ++f)
      opre[(size_t)(b * L_ + gi) * C_ + h * DV + f * 16 + colw] = O[f][r] * inv;
  }
}

// ---------------------------------------------------------------------------
// Final GroupNorm apply (32 groups over C=512 -> 16 ch/group)
// ---------------------------------------------------------------------------
__global__ void k_gn_apply(const float* __restrict__ x,
                           const float* __restrict__ stats,
                           const float* __restrict__ scale,
                           const float* __restrict__ bias,
                           float* __restrict__ out) {
  const int idx = blockIdx.x * blockDim.x + threadIdx.x;
  if (idx >= B_ * L_ * C_) return;
  const int c = idx & (C_ - 1);
  const int b = idx >> 19;
  const int g = c >> 4;
  const float mean = stats[(b * 32 + g) * 2 + 0];
  const float rstd = stats[(b * 32 + g) * 2 + 1];
  out[idx] = (x[idx] - mean) * rstd * scale[c] + bias[c];
}

// ---------------------------------------------------------------------------
extern "C" void kernel_launch(void* const* d_in, const int* in_sizes, int n_in,
                              void* d_out, int out_size, void* d_ws, size_t ws_size,
                              hipStream_t stream) {
  const float* x    = (const float*)d_in[0];
  const float* Wqkv = (const float*)d_in[1];
  const float* gqs  = (const float*)d_in[2];
  const float* gqb  = (const float*)d_in[3];
  const float* E    = (const float*)d_in[4];
  const float* gos  = (const float*)d_in[5];
  const float* gob  = (const float*)d_in[6];
  float* out = (float*)d_out;

  char* w = (char*)d_ws;
  auto alloc = [&](size_t bytes) {
    char* p = w; w += (bytes + 255) & ~(size_t)255; return p;
  };
  unsigned short* pA  = (unsigned short*)alloc((size_t)B_ * L_ * C_  * 2);
  unsigned short* pB  = (unsigned short*)alloc((size_t)C_ * C2_      * 2);
  float*          qkv = (float*)         alloc((size_t)B_ * L_ * C2_ * 4);
  float*          st1 = (float*)         alloc((size_t)B_ * 32 * 2   * 4);
  unsigned short* qT  = (unsigned short*)alloc((size_t)B_ * H_ * L_ * DQ * 2);
  unsigned short* kT  = (unsigned short*)alloc((size_t)B_ * H_ * L_ * DQ * 2);
  unsigned short* vP  = (unsigned short*)alloc((size_t)B_ * H_ * L_ * DV * 2);
  float*          opre= (float*)         alloc((size_t)B_ * L_ * C_  * 4);
  float*          st2 = (float*)         alloc((size_t)B_ * 32 * 2   * 4);

  k_pack_a<<<(B_ * L_ * C_ + 255) / 256, 256, 0, stream>>>(x, pA);
  k_pack_b<<<(C_ * C2_ + 255) / 256, 256, 0, stream>>>(Wqkv, pB);
  k_qkv_gemm<<<512, 256, 0, stream>>>(pA, pB, qkv);
  k_gn_stats<<<B_ * 32, 256, 0, stream>>>(qkv, st1, C2_, 32);
  k_qkv_norm_split<<<(B_ * L_ * C2_ + 255) / 256, 256, 0, stream>>>(
      qkv, st1, gqs, gqb, qT, kT, vP);
  k_attention<<<dim3(L_ / 16, H_, B_), 32, 0, stream>>>(qT, kT, vP, E, opre);
  k_gn_stats<<<B_ * 32, 256, 0, stream>>>(opre, st2, C_, 16);
  k_gn_apply<<<(B_ * L_ * C_ + 255) / 256, 256, 0, stream>>>(opre, st2, gos, gob, out);
}
